// HeteroGraphTransformer_70944269795722
// MI455X (gfx1250) — compile-verified, matrix operand order checked by reading, not run on hardware
//
#include <hip/hip_runtime.h>
#include <hip/hip_bf16.h>

// ---------------------------------------------------------------------------
// HGT forward on gfx1250 (CDNA5). All dense math (input proj, K/Q/V proj,
// per-head relation transforms via block-diagonal weights, output proj) runs
// through v_wmma_f32_16x16x32_f16 with ds_load_b128 fragment loads.
// Sparse edge softmax/aggregation uses VALU + global float atomics.
// ---------------------------------------------------------------------------

typedef __attribute__((ext_vector_type(16))) _Float16 v16h;
typedef __attribute__((ext_vector_type(8)))  _Float16 v8h;
typedef __attribute__((ext_vector_type(8)))  float    v8f;

#define NUM_T   3
#define NUM_R   4
#define NUM_L   2
#define HEADS   8
#define HDIM    16
#define OUTF    128      // HEADS*HDIM
#define NT      30000
#define NEDGE   200000
#define MTILES  3        // row tiles of 16 per block -> 48 rows/block
#define ROWS_PB (16 * MTILES)

struct FragPair { v8h lo, hi; };   // two contiguous 8-half runs -> v16h operand

__device__ __forceinline__ float gelu_exact(float x) {
    return 0.5f * x * (1.0f + erff(x * 0.70710678118654752f));
}

__device__ __forceinline__ void atomic_max_f32(float* addr, float val) {
    int* ai = (int*)addr;
    int old = __float_as_int(*addr);
    while (true) {
        float cur = __int_as_float(old);
        if (cur >= val) break;
        int assumed = old;
        old = atomicCAS(ai, assumed, __float_as_int(val));
        if (old == assumed) break;
    }
}

// ---------------------------------------------------------------------------
// WMMA GEMM: per block, C[48 x 128] = act(A[48x128] @ W[128x128] + bias).
// 8 waves; wave w owns a 16-wide column strip and 3 row tiles (12 WMMAs).
// W is staged transposed in LDS so both A and B fragments are two aligned
// 16-byte runs -> ds_load_b128.
// flags: bit0 = pre-GELU on A, bit1 = post-ReLU.
// ---------------------------------------------------------------------------
__global__ __launch_bounds__(256) void gemm128_wmma(
    const float* __restrict__ A, const float* __restrict__ W,
    const float* __restrict__ bias, const float* __restrict__ skip_in,
    const float* __restrict__ skip_logit, float* __restrict__ C, int flags)
{
    __shared__ alignas(16) _Float16 Alds[ROWS_PB][136];  // 272B pitch (16B mult)
    __shared__ alignas(16) _Float16 Wt[128][136];        // transposed: Wt[n][k]

    const int tid  = threadIdx.x;
    const int row0 = blockIdx.x * ROWS_PB;
    const bool pre_gelu  = (flags & 1) != 0;
    const bool post_relu = (flags & 2) != 0;

    __builtin_prefetch(W, 0, 3);  // global_prefetch_b8 of the weight panel

    // Stage A tile (48x128), fp32 -> f16, optional pre-activation.
    for (int i = tid; i < ROWS_PB * 128; i += 256) {
        int r = i >> 7, c = i & 127;
        float x = A[(size_t)(row0 + r) * OUTF + c];
        if (pre_gelu) x = gelu_exact(x);
        Alds[r][c] = (_Float16)x;
    }
    // Stage W (128x128) transposed, fp32 -> f16 (coalesced global reads).
    for (int i = tid; i < 128 * 128; i += 256) {
        int r = i >> 7, c = i & 127;
        Wt[c][r] = (_Float16)W[i];
    }
    __syncthreads();

    const int lane  = tid & 31;
    const int wave  = tid >> 5;       // 8 waves -> 8 column strips of 16
    const int nlow  = lane & 15;
    const int khalf = lane >> 4;      // lane[4] selects K half per ISA layout
    const int ncol  = wave * 16 + nlow;

    v8f acc[MTILES] = {};
#pragma unroll
    for (int kk = 0; kk < 4; ++kk) {  // K = 128 = 4 x 32
        const int k0 = kk * 32 + khalf * 8;   // halves; 16B aligned
        FragPair bf;
        bf.lo = *(const v8h*)&Wt[ncol][k0];
        bf.hi = *(const v8h*)&Wt[ncol][k0 + 16];
        const v16h b = __builtin_bit_cast(v16h, bf);
#pragma unroll
        for (int mt = 0; mt < MTILES; ++mt) {
            FragPair af;
            af.lo = *(const v8h*)&Alds[mt * 16 + nlow][k0];
            af.hi = *(const v8h*)&Alds[mt * 16 + nlow][k0 + 16];
            const v16h a = __builtin_bit_cast(v16h, af);
            acc[mt] = __builtin_amdgcn_wmma_f32_16x16x32_f16(
                false, a, false, b, (short)0, acc[mt], false, false);
        }
    }

    const float bv = bias ? bias[ncol] : 0.0f;
    float sig = 0.0f;
    if (skip_in) sig = 1.0f / (1.0f + __expf(-skip_logit[0]));
#pragma unroll
    for (int mt = 0; mt < MTILES; ++mt) {
#pragma unroll
        for (int v = 0; v < 8; ++v) {
            int m = row0 + mt * 16 + v + 8 * khalf;  // f32 C layout: M=v+8*lane[4]
            float val = acc[mt][v] + bv;
            if (post_relu) val = fmaxf(val, 0.0f);
            size_t idx = (size_t)m * OUTF + ncol;
            if (skip_in) val = sig * val + (1.0f - sig) * skip_in[idx];
            C[idx] = val;
        }
    }
}

// Expand per-head relation weights [H,16,16] into a 128x128 block-diagonal
// matrix so the einsum('nhd,hde->nhe') becomes a standard WMMA GEMM.
__global__ void expand_blockdiag(const float* __restrict__ rel, float* __restrict__ Wbd)
{
    int i = blockIdx.x * blockDim.x + threadIdx.x;   // 16384 elements
    if (i >= 128 * 128) return;
    int r = i >> 7, c = i & 127;
    int hr = r >> 4, dd = r & 15;
    int hc = c >> 4, ee = c & 15;
    Wbd[i] = (hr == hc) ? rel[(hr * 16 + dd) * 16 + ee] : 0.0f;
}

__global__ void fill_f32(float* __restrict__ p, float v, int n)
{
    int i = blockIdx.x * blockDim.x + threadIdx.x;
    if (i < n) p[i] = v;
}

// Pass 1: logits[e,h] = (q[dst] . k_rel[src]) * p_rel[h] / sqrt(D); seg max.
__global__ void edge_logits_max(
    const int* __restrict__ src, const int* __restrict__ dst,
    const float* __restrict__ q, const float* __restrict__ krel,
    const float* __restrict__ prel, float* __restrict__ logits,
    float* __restrict__ mmax)
{
    int i = blockIdx.x * blockDim.x + threadIdx.x;
    if (i >= NEDGE * HEADS) return;
    int e = i >> 3, h = i & 7;
    int s = src[e], d = dst[e];
    const float* qp = q    + (size_t)d * OUTF + h * HDIM;
    const float* kp = krel + (size_t)s * OUTF + h * HDIM;
    float acc = 0.0f;
#pragma unroll
    for (int j = 0; j < HDIM; ++j) acc += qp[j] * kp[j];
    acc *= prel[h] * 0.25f;          // 1/sqrt(16)
    logits[i] = acc;
    atomic_max_f32(&mmax[d * HEADS + h], acc);
}

// Pass 2: e = exp(logit - m[dst]); seg sum. (logits overwritten with exp)
__global__ void edge_exp_sum(
    const int* __restrict__ dst, float* __restrict__ logits,
    const float* __restrict__ mmax, float* __restrict__ ssum)
{
    int i = blockIdx.x * blockDim.x + threadIdx.x;
    if (i >= NEDGE * HEADS) return;
    int e = i >> 3, h = i & 7;
    int d = dst[e];
    float m = mmax[d * HEADS + h];
    if (!isfinite(m)) m = 0.0f;
    float ex = __expf(logits[i] - m);
    logits[i] = ex;
    atomicAdd(&ssum[d * HEADS + h], ex);
}

// Pass 3: alpha = e/(sum+eps); agg[dst] += alpha * v_rel[src]  (16 wide)
__global__ void edge_scatter(
    const int* __restrict__ src, const int* __restrict__ dst,
    const float* __restrict__ logits, const float* __restrict__ ssum,
    const float* __restrict__ vrel, float* __restrict__ agg)
{
    int i = blockIdx.x * blockDim.x + threadIdx.x;
    if (i >= NEDGE * HEADS) return;
    int e = i >> 3, h = i & 7;
    int s = src[e], d = dst[e];
    float alpha = logits[i] / (ssum[d * HEADS + h] + 1e-16f);
    const float* vp = vrel + (size_t)s * OUTF + h * HDIM;
    float*       ap = agg  + (size_t)d * OUTF + h * HDIM;
#pragma unroll
    for (int j = 0; j < HDIM; ++j) atomicAdd(&ap[j], vp[j] * alpha);
}

// ---------------------------------------------------------------------------
extern "C" void kernel_launch(void* const* d_in, const int* in_sizes, int n_in,
                              void* d_out, int out_size, void* d_ws, size_t ws_size,
                              hipStream_t stream)
{
    (void)in_sizes; (void)n_in; (void)out_size; (void)ws_size;

    const float* x_t[NUM_T] = {(const float*)d_in[0], (const float*)d_in[1],
                               (const float*)d_in[2]};
    const float* lin_w = (const float*)d_in[4];
    const float* lin_b = (const float*)d_in[5];
    const float* k_w   = (const float*)d_in[6];
    const float* k_b   = (const float*)d_in[7];
    const float* q_w   = (const float*)d_in[8];
    const float* q_b   = (const float*)d_in[9];
    const float* v_w   = (const float*)d_in[10];
    const float* v_b   = (const float*)d_in[11];
    const float* a_w   = (const float*)d_in[12];
    const float* a_b   = (const float*)d_in[13];
    const float* skip  = (const float*)d_in[14];
    const float* a_rel = (const float*)d_in[15];
    const float* m_rel = (const float*)d_in[16];
    const float* p_rel = (const float*)d_in[17];
    const int*   ei    = (const int*)d_in[18];
    // d_in[19] node_type_argmax == repeat(arange(3), NT): output scatter is
    // plain per-type concatenation, handled by writing d_out segments directly.

    float* out = (float*)d_out;
    float* ws  = (float*)d_ws;

    const size_t NF = (size_t)NT * OUTF;   // floats per node-type matrix
    float *hsA[NUM_T], *hsB[NUM_T], *kb[NUM_T], *qb[NUM_T], *vb[NUM_T], *agg[NUM_T];
    for (int t = 0; t < NUM_T; ++t) {
        hsA[t] = ws + (size_t)(0  + t) * NF;
        hsB[t] = ws + (size_t)(3  + t) * NF;
        kb[t]  = ws + (size_t)(6  + t) * NF;
        qb[t]  = ws + (size_t)(9  + t) * NF;
        vb[t]  = ws + (size_t)(12 + t) * NF;
        agg[t] = ws + (size_t)(17 + t) * NF;
    }
    float* krel   = ws + 15 * NF;
    float* vrel   = ws + 16 * NF;
    float* logits = ws + 20 * NF;                       // NEDGE*HEADS floats
    float* mmax   = logits + (size_t)NEDGE * HEADS;     // NT*HEADS
    float* ssum   = mmax   + (size_t)NT * HEADS;        // NT*HEADS
    float* Wbd    = ssum   + (size_t)NT * HEADS;        // 128*128

    const dim3 gGemm(NT / ROWS_PB), bGemm(256);         // 30000/48 = 625
    const int  EH  = NEDGE * HEADS;
    const int  gEH = (EH + 255) / 256;
    const int  gNH = (NT * HEADS + 255) / 256;
    const int  gNF = (int)((NF + 255) / 256);

    // ---- input projection + ReLU ----
    for (int t = 0; t < NUM_T; ++t)
        gemm128_wmma<<<gGemm, bGemm, 0, stream>>>(
            x_t[t], lin_w + (size_t)t * 16384, lin_b + (size_t)t * 128,
            nullptr, nullptr, hsA[t], /*post_relu*/ 2);

    const int EDGE_SRC[NUM_R] = {0, 0, 1, 0};
    const int EDGE_DST[NUM_R] = {0, 1, 0, 2};

    float* cur[NUM_T] = {hsA[0], hsA[1], hsA[2]};
    float* nxt[NUM_T] = {hsB[0], hsB[1], hsB[2]};

    for (int l = 0; l < NUM_L; ++l) {
        // ---- K/Q/V projections + zero aggregation buffers ----
        for (int t = 0; t < NUM_T; ++t) {
            size_t wOff = (size_t)(l * NUM_T + t) * 16384;
            size_t bOff = (size_t)(l * NUM_T + t) * 128;
            gemm128_wmma<<<gGemm, bGemm, 0, stream>>>(
                cur[t], k_w + wOff, k_b + bOff, nullptr, nullptr, kb[t], 0);
            gemm128_wmma<<<gGemm, bGemm, 0, stream>>>(
                cur[t], q_w + wOff, q_b + bOff, nullptr, nullptr, qb[t], 0);
            gemm128_wmma<<<gGemm, bGemm, 0, stream>>>(
                cur[t], v_w + wOff, v_b + bOff, nullptr, nullptr, vb[t], 0);
            fill_f32<<<gNF, 256, 0, stream>>>(agg[t], 0.0f, (int)NF);
        }

        // ---- relations: rel transform (block-diag WMMA GEMM) + softmax agg ----
        for (int r = 0; r < NUM_R; ++r) {
            const int s = EDGE_SRC[r], d = EDGE_DST[r];
            size_t relOff = (size_t)(l * NUM_R + r) * HEADS * HDIM * HDIM;

            expand_blockdiag<<<64, 256, 0, stream>>>(a_rel + relOff, Wbd);
            gemm128_wmma<<<gGemm, bGemm, 0, stream>>>(
                kb[s], Wbd, nullptr, nullptr, nullptr, krel, 0);
            expand_blockdiag<<<64, 256, 0, stream>>>(m_rel + relOff, Wbd);
            gemm128_wmma<<<gGemm, bGemm, 0, stream>>>(
                vb[s], Wbd, nullptr, nullptr, nullptr, vrel, 0);

            fill_f32<<<gNH, 256, 0, stream>>>(mmax, -3.0e38f, NT * HEADS);
            fill_f32<<<gNH, 256, 0, stream>>>(ssum, 0.0f, NT * HEADS);

            const int* srcIdx = ei + ((size_t)r * 2 + 0) * NEDGE;
            const int* dstIdx = ei + ((size_t)r * 2 + 1) * NEDGE;
            const float* prl  = p_rel + (size_t)(l * NUM_R + r) * HEADS;

            edge_logits_max<<<gEH, 256, 0, stream>>>(srcIdx, dstIdx, qb[d], krel,
                                                     prl, logits, mmax);
            edge_exp_sum<<<gEH, 256, 0, stream>>>(dstIdx, logits, mmax, ssum);
            edge_scatter<<<gEH, 256, 0, stream>>>(srcIdx, dstIdx, logits, ssum,
                                                  vrel, agg[d]);
        }

        // ---- output projection: GELU(agg) @ a_w + a_b, sigmoid-skip blend ----
        for (int t = 0; t < NUM_T; ++t) {
            size_t wOff = (size_t)(l * NUM_T + t) * 16384;
            size_t bOff = (size_t)(l * NUM_T + t) * 128;
            float* dstp = (l == NUM_L - 1) ? (out + (size_t)t * NF) : nxt[t];
            gemm128_wmma<<<gGemm, bGemm, 0, stream>>>(
                agg[t], a_w + wOff, a_b + bOff, cur[t],
                skip + (size_t)(l * NUM_T + t), dstp, /*pre_gelu*/ 1);
        }
        for (int t = 0; t < NUM_T; ++t) { float* tmp = cur[t]; cur[t] = nxt[t]; nxt[t] = tmp; }
    }
}
